// MultiHeadSelfAttention_11476152615324
// MI455X (gfx1250) — matrix operand order repaired and checked by reading, not compiled, not run on hardware
//
#include <hip/hip_runtime.h>

// -------- types for CDNA5 WMMA (wave32) --------
typedef __bf16 bf16;
typedef __attribute__((ext_vector_type(16))) __bf16 v16bf;
typedef __attribute__((ext_vector_type(8)))  __bf16 v8bf;
typedef __attribute__((ext_vector_type(8)))  float  v8f;

union V16U { v16bf v; v8bf h[2]; };

static __device__ inline v8f wmma_bf16(v16bf a, v16bf b, v8f c) {
  // D = A(16x32) * B(32x16) + C(16x16 f32)
  return __builtin_amdgcn_wmma_f32_16x16x32_bf16(false, a, false, b, (short)0, c,
                                                 false, false);
}

// A-fragment (16x32 bf16): lane holds row m = lane&15; per lane 16 contiguous
// elements at local-K = kh*16 .. kh*16+15 (kh = lane>>4).
static __device__ inline v16bf load_frag_a(const bf16* rowbase, int koff, int kh) {
  V16U u;
  u.h[0] = *(const v8bf*)(rowbase + koff + kh * 16);
  u.h[1] = *(const v8bf*)(rowbase + koff + kh * 16 + 8);
  return u.v;
}

// ======================= fp32 -> bf16 conversion =======================
__global__ void cvt_f32_bf16(const float* __restrict__ in, bf16* __restrict__ out, int n) {
  int i = blockIdx.x * blockDim.x + threadIdx.x;
  if (i < n) out[i] = (bf16)in[i];
}

// ============ W[k][n] fp32 -> Wt[n][k] bf16  (LDS tiled transpose) ============
__global__ void transpose_w(const float* __restrict__ W, bf16* __restrict__ Wt) {
  __shared__ bf16 tile[32][33];
  const int bx = blockIdx.x & 31;   // k-tile index
  const int by = blockIdx.x >> 5;   // n-tile index
  const int tx = threadIdx.x & 31;
  const int ty = threadIdx.x >> 5;  // 0..7
#pragma unroll
  for (int j = 0; j < 4; ++j) {
    int k = bx * 32 + ty + j * 8;
    int n = by * 32 + tx;
    tile[ty + j * 8][tx] = (bf16)W[k * 1024 + n];   // coalesced read over n
  }
  __syncthreads();
#pragma unroll
  for (int j = 0; j < 4; ++j) {
    int n = by * 32 + ty + j * 8;
    int k = bx * 32 + tx;
    Wt[n * 1024 + k] = tile[tx][ty + j * 8];        // coalesced write over k
  }
}

// ======================= projection GEMM: [8192x1024] @ W -> Q/K/V =======================
// One wave -> 64x64 output tile (4 A-frags x 4 B-frags -> 16 WMMA per k-step).
// __launch_bounds__(128,1): full register file, no spills (acc=128 VGPRs).
// TRANSPOSE_OUT==0: store [b][h][s][d];  ==1: store [b][h][d][s] (V).
template <int TRANSPOSE_OUT>
__global__ void __launch_bounds__(128, 1)
proj_kernel(const bf16* __restrict__ X, const bf16* __restrict__ Wt,
            bf16* __restrict__ Out, float scale) {
  const int lane = threadIdx.x & 31;
  const int wave = threadIdx.x >> 5;
  const int wid  = blockIdx.x * 4 + wave;  // 0..2047
  const int mt = wid >> 4;                 // 0..127 (64-row tiles)
  const int nt = wid & 15;                 // 0..15  (64-col tiles)
  const int m0 = mt * 64, n0 = nt * 64;
  const int kh = lane >> 4;
  const int ml = lane & 15;

  v8f acc[4][4] = {};
  const bf16* arow[4];
#pragma unroll
  for (int qm = 0; qm < 4; ++qm)
    arow[qm] = X + (size_t)(m0 + qm * 16 + ml) * 1024;

  for (int k0 = 0; k0 < 1024; k0 += 32) {
    v16bf a[4];
#pragma unroll
    for (int qm = 0; qm < 4; ++qm) a[qm] = load_frag_a(arow[qm], k0, kh);
#pragma unroll
    for (int dt = 0; dt < 4; ++dt) {
      v16bf bfr = *(const v16bf*)(Wt + (size_t)(n0 + dt * 16 + ml) * 1024 + k0 + kh * 16);
#pragma unroll
      for (int qm = 0; qm < 4; ++qm)
        acc[qm][dt] = wmma_bf16(a[qm], bfr, acc[qm][dt]);
    }
  }
#pragma unroll
  for (int qm = 0; qm < 4; ++qm) {
#pragma unroll
    for (int dt = 0; dt < 4; ++dt) {
#pragma unroll
      for (int r = 0; r < 8; ++r) {
        int M = m0 + qm * 16 + r + 8 * kh;    // global row (b*2048+s)
        int n = n0 + dt * 16 + ml;            // output feature
        int b = M >> 11, s = M & 2047;
        int h = n >> 6, d = n & 63;
        bf16 val = (bf16)(acc[qm][dt][r] * scale);
        if (TRANSPOSE_OUT == 0)
          Out[(size_t)(((b * 16 + h) * 2048 + s)) * 64 + d] = val;
        else
          Out[(size_t)(((b * 16 + h) * 64 + d)) * 2048 + s] = val;
      }
    }
  }
}

// ======================= flash attention =======================
// One wave -> one (b,h, 32-row q-tile): K/V fragments shared by two q sub-tiles.
// Online softmax in base-2 domain (log2(e) folded into Q projection scale).
__global__ void __launch_bounds__(128, 1)
attn_kernel(const bf16* __restrict__ Q, const bf16* __restrict__ K,
            const bf16* __restrict__ Vt, bf16* __restrict__ Ctx) {
  __shared__ alignas(16) bf16 Pl[4][32][32];   // per-wave P tile (32 q x 32 k)
  const int lane = threadIdx.x & 31;
  const int wave = threadIdx.x >> 5;
  const int wid  = blockIdx.x * 4 + wave;      // 0..4095
  const int qt = wid & 63;                     // 64 q-tiles of 32 rows
  const int bh = wid >> 6;                     // b*16+h, 0..63
  const int q0 = qt * 32;
  const int kh = lane >> 4;
  const int ml = lane & 15;
  const int b = bh >> 4, h = bh & 15;

  v16bf aQ[2][2];
#pragma unroll
  for (int qm = 0; qm < 2; ++qm) {
    const bf16* Qbase = Q + (size_t)(bh * 2048 + q0 + qm * 16 + ml) * 64;
    aQ[qm][0] = load_frag_a(Qbase, 0, kh);
    aQ[qm][1] = load_frag_a(Qbase, 32, kh);
  }

  v8f acc[2][4] = {};
  float mrow[2][8], lrow[2][8];
#pragma unroll
  for (int qm = 0; qm < 2; ++qm)
#pragma unroll
    for (int r = 0; r < 8; ++r) { mrow[qm][r] = -3.0e38f; lrow[qm][r] = 0.0f; }

  const bf16* Kbh = K + (size_t)bh * 2048 * 64;
  const bf16* Vbh = Vt + (size_t)bh * 64 * 2048;
  const int rl = kh * 16 + ml;                 // 0..31 across the wave

  for (int kt = 0; kt < 2048; kt += 32) {
    if (kt + 32 < 2048) {
      // next K tile: rows kt+32..kt+63 fully covered (rl spans 0..31)
      __builtin_prefetch(Kbh + (size_t)(kt + 32 + rl) * 64, 0, 3);
      // next V columns: rows d=0..31 and d=32..63
      __builtin_prefetch(Vbh + (size_t)rl * 2048 + kt + 32, 0, 3);
      __builtin_prefetch(Vbh + (size_t)(32 + rl) * 2048 + kt + 32, 0, 3);
    }
    // ---- scores S = Q * K^T for 32 keys (2 q sub-tiles x 2 key sub-tiles) ----
    v16bf bK[2][2];
#pragma unroll
    for (int ct = 0; ct < 2; ++ct) {
      const bf16* kp = Kbh + (size_t)(kt + ct * 16 + ml) * 64;
      bK[ct][0] = *(const v16bf*)(kp + kh * 16);
      bK[ct][1] = *(const v16bf*)(kp + 32 + kh * 16);
    }
    v8f S[2][2];
#pragma unroll
    for (int qm = 0; qm < 2; ++qm)
#pragma unroll
      for (int ct = 0; ct < 2; ++ct) {
        v8f z = {};
        z = wmma_bf16(aQ[qm][0], bK[ct][0], z);
        S[qm][ct] = wmma_bf16(aQ[qm][1], bK[ct][1], z);
      }

    // ---- online softmax (base-2): row stats per C-fragment VGPR slot ----
#pragma unroll
    for (int qm = 0; qm < 2; ++qm) {
#pragma unroll
      for (int r = 0; r < 8; ++r) {
        float tmax = fmaxf(S[qm][0][r], S[qm][1][r]);
        tmax = fmaxf(tmax, __shfl_xor(tmax, 1));
        tmax = fmaxf(tmax, __shfl_xor(tmax, 2));
        tmax = fmaxf(tmax, __shfl_xor(tmax, 4));
        tmax = fmaxf(tmax, __shfl_xor(tmax, 8));  // row max over 16-lane half
        float mnew = fmaxf(mrow[qm][r], tmax);
        float corr = exp2f(mrow[qm][r] - mnew);
        mrow[qm][r] = mnew;
        float p0 = exp2f(S[qm][0][r] - mnew);
        float p1 = exp2f(S[qm][1][r] - mnew);
        float rsum = p0 + p1;
        rsum += __shfl_xor(rsum, 1);
        rsum += __shfl_xor(rsum, 2);
        rsum += __shfl_xor(rsum, 4);
        rsum += __shfl_xor(rsum, 8);
        lrow[qm][r] = lrow[qm][r] * corr + rsum;
        acc[qm][0][r] *= corr; acc[qm][1][r] *= corr;
        acc[qm][2][r] *= corr; acc[qm][3][r] *= corr;
        int M = qm * 16 + r + 8 * kh;
        Pl[wave][M][ml]      = (bf16)p0;
        Pl[wave][M][16 + ml] = (bf16)p1;
      }
    }

    // ---- re-load P as A-fragments (2 x 16x32) from LDS ----
    v16bf aP[2];
#pragma unroll
    for (int qm = 0; qm < 2; ++qm) {
      V16U up;
      up.h[0] = *(const v8bf*)(&Pl[wave][qm * 16 + ml][kh * 16]);
      up.h[1] = *(const v8bf*)(&Pl[wave][qm * 16 + ml][kh * 16 + 8]);
      aP[qm] = up.v;
    }

    // ---- context += P * V (V transposed: K-contiguous in keys) ----
    v16bf bV[4];
#pragma unroll
    for (int dt = 0; dt < 4; ++dt)
      bV[dt] = *(const v16bf*)(Vbh + (size_t)(dt * 16 + ml) * 2048 + kt + kh * 16);
#pragma unroll
    for (int qm = 0; qm < 2; ++qm)
#pragma unroll
      for (int dt = 0; dt < 4; ++dt)
        acc[qm][dt] = wmma_bf16(aP[qm], bV[dt], acc[qm][dt]);
  }

  // ---- normalize and store context as [b][s][H*Dh] bf16 ----
#pragma unroll
  for (int qm = 0; qm < 2; ++qm) {
#pragma unroll
    for (int r = 0; r < 8; ++r) {
      float inv = 1.0f / lrow[qm][r];
      int M = q0 + qm * 16 + r + 8 * kh;
      bf16* crow = Ctx + (size_t)(b * 2048 + M) * 1024 + h * 64;
#pragma unroll
      for (int dt = 0; dt < 4; ++dt)
        crow[dt * 16 + ml] = (bf16)(acc[qm][dt][r] * inv);
    }
  }
}

// ======================= output GEMM: ctx @ Wo + bo -> f32 =======================
// One wave -> 64x64 output tile, same blocking as proj_kernel.
__global__ void __launch_bounds__(128, 1)
out_gemm_kernel(const bf16* __restrict__ Ctx, const bf16* __restrict__ Wot,
                const float* __restrict__ bo, float* __restrict__ Out) {
  const int lane = threadIdx.x & 31;
  const int wave = threadIdx.x >> 5;
  const int wid  = blockIdx.x * 4 + wave;  // 0..2047
  const int mt = wid >> 4;
  const int nt = wid & 15;
  const int m0 = mt * 64, n0 = nt * 64;
  const int kh = lane >> 4;
  const int ml = lane & 15;

  v8f acc[4][4] = {};
  const bf16* arow[4];
#pragma unroll
  for (int qm = 0; qm < 4; ++qm)
    arow[qm] = Ctx + (size_t)(m0 + qm * 16 + ml) * 1024;

  for (int k0 = 0; k0 < 1024; k0 += 32) {
    v16bf a[4];
#pragma unroll
    for (int qm = 0; qm < 4; ++qm) a[qm] = load_frag_a(arow[qm], k0, kh);
#pragma unroll
    for (int dt = 0; dt < 4; ++dt) {
      v16bf bfr = *(const v16bf*)(Wot + (size_t)(n0 + dt * 16 + ml) * 1024 + k0 + kh * 16);
#pragma unroll
      for (int qm = 0; qm < 4; ++qm)
        acc[qm][dt] = wmma_bf16(a[qm], bfr, acc[qm][dt]);
    }
  }
#pragma unroll
  for (int dt = 0; dt < 4; ++dt) {
    int n = n0 + dt * 16 + ml;
    float bias = bo[n];
#pragma unroll
    for (int qm = 0; qm < 4; ++qm) {
#pragma unroll
      for (int r = 0; r < 8; ++r) {
        int M = m0 + qm * 16 + r + 8 * kh;
        Out[(size_t)M * 1024 + n] = acc[qm][dt][r] + bias;
      }
    }
  }
}

// ======================= host-side launch =======================
extern "C" void kernel_launch(void* const* d_in, const int* in_sizes, int n_in,
                              void* d_out, int out_size, void* d_ws, size_t ws_size,
                              hipStream_t stream) {
  const float* x  = (const float*)d_in[0];
  const float* Wq = (const float*)d_in[1];
  const float* Wk = (const float*)d_in[2];
  const float* Wv = (const float*)d_in[3];
  const float* Wo = (const float*)d_in[4];
  const float* bo = (const float*)d_in[5];
  float* out = (float*)d_out;
  char* ws = (char*)d_ws;

  const size_t MB = 1024ull * 1024ull;
  bf16* xb  = (bf16*)(ws);             // 16 MB : x in bf16          [8192][1024]
  bf16* Wtq = (bf16*)(ws + 16 * MB);   //  2 MB : Wq^T bf16          [n][k]
  bf16* Wtk = (bf16*)(ws + 18 * MB);
  bf16* Wtv = (bf16*)(ws + 20 * MB);
  bf16* Wto = (bf16*)(ws + 22 * MB);
  bf16* Qb  = (bf16*)(ws + 24 * MB);   // 16 MB : Q bf16 [b][h][s][d] (scaled, base-2)
  bf16* Kb  = (bf16*)(ws + 40 * MB);   // 16 MB : K bf16 [b][h][s][d]
  bf16* Vtb = (bf16*)(ws + 56 * MB);   // 16 MB : V^T bf16 [b][h][d][s]
  bf16* Ctx = (bf16*)(ws + 72 * MB);   // 16 MB : context bf16 [b][s][1024]

  cvt_f32_bf16<<<32768, 256, 0, stream>>>(x, xb, 4 * 2048 * 1024);
  transpose_w<<<1024, 256, 0, stream>>>(Wq, Wtq);
  transpose_w<<<1024, 256, 0, stream>>>(Wk, Wtk);
  transpose_w<<<1024, 256, 0, stream>>>(Wv, Wtv);
  transpose_w<<<1024, 256, 0, stream>>>(Wo, Wto);

  // DIM_HEAD^-0.5 * log2(e): scores land in base-2 domain for exp2-softmax
  const float scale = 0.125f * 1.44269504088896f;
  proj_kernel<0><<<512, 128, 0, stream>>>(xb, Wtq, Qb, scale);
  proj_kernel<0><<<512, 128, 0, stream>>>(xb, Wtk, Kb, 1.0f);
  proj_kernel<1><<<512, 128, 0, stream>>>(xb, Wtv, Vtb, 1.0f);

  attn_kernel<<<1024, 128, 0, stream>>>(Qb, Kb, Vtb, Ctx);

  out_gemm_kernel<<<512, 128, 0, stream>>>(Ctx, Wto, bo, out);
}
